// NodeModel_20667382628990
// MI455X (gfx1250) — compile-verified, matrix operand order checked by reading, not run on hardware
//
#include <hip/hip_runtime.h>

typedef __attribute__((ext_vector_type(16))) _Float16 v16h;
typedef __attribute__((ext_vector_type(8)))  _Float16 v8h;
typedef __attribute__((ext_vector_type(8)))  float    v8f;
typedef __attribute__((ext_vector_type(4)))  float    v4f;

#define MSG_STRIDE 264   // 256 + 8 f16 pad
#define W1_STRIDE  264   // 256 + 8
#define Z_STRIDE   392   // 384 + 8
#define W2A_STRIDE 392   // 384 + 8
#define W2B_STRIDE 264   // 256 + 8

// Build a 16x32 f16 A/B WMMA fragment for one lane from a row-contiguous
// f16 array. Per the CDNA5 16-bit A layout: lanes 0-15 hold K = kbase..kbase+7
// and kbase+16..kbase+23 (kbase = k0), lanes 16-31 the same with kbase = k0+8.
__device__ __forceinline__ v16h load_frag(const _Float16* base, int kbase) {
    v8h lo = *(const v8h*)(base + kbase);
    v8h hi = *(const v8h*)(base + kbase + 16);
    return __builtin_shufflevector(lo, hi, 0,1,2,3,4,5,6,7,8,9,10,11,12,13,14,15);
}

__device__ __forceinline__ float elu_f(float v) {
    return v > 0.f ? v : (__expf(v) - 1.f);
}

#define WMMA_F16(A, B, C) \
    __builtin_amdgcn_wmma_f32_16x16x32_f16(false, (A), false, (B), (short)0, (C), false, false)

// ---------------------------------------------------------------- weights prep
__global__ void convert_weights_kernel(const float* __restrict__ W1a,
                                       const float* __restrict__ W1b,
                                       const float* __restrict__ W2a,
                                       const float* __restrict__ W2b,
                                       _Float16* __restrict__ w1aT,
                                       _Float16* __restrict__ w1bT,
                                       _Float16* __restrict__ w2aT,
                                       _Float16* __restrict__ w2bT) {
    int i0 = blockIdx.x * blockDim.x + threadIdx.x;
    int stride = gridDim.x * blockDim.x;
    // W1a [256 in][256 out] -> w1aT[out][in]
    for (int i = i0; i < 256 * 256; i += stride) {
        int n = i >> 8, k = i & 255;
        w1aT[i] = (_Float16)W1a[k * 256 + n];
    }
    for (int i = i0; i < 256 * 256; i += stride) {
        int n = i >> 8, k = i & 255;
        w1bT[i] = (_Float16)W1b[k * 256 + n];
    }
    // W2a [384 in][256 out] -> w2aT[out=256][in=384]
    for (int i = i0; i < 256 * 384; i += stride) {
        int n = i / 384, k = i - n * 384;
        w2aT[i] = (_Float16)W2a[k * 256 + n];
    }
    // W2b [256 in][128 out] -> w2bT[out=128][in=256]
    for (int i = i0; i < 128 * 256; i += stride) {
        int n = i >> 8, k = i & 255;
        w2bT[i] = (_Float16)W2b[k * 128 + n];
    }
}

__global__ void zero_kernel(float* __restrict__ p, long n) {
    long i = (long)blockIdx.x * blockDim.x + threadIdx.x;
    long stride = (long)gridDim.x * blockDim.x;
    for (; i < n; i += stride) p[i] = 0.f;
}

// ---------------------------------------------------------------- edge MLP
// Per 32-edge tile: msg = [x[row] || edge_attr]  (K=256)
// h = (ELU(msg @ W1a + b1a)) @ W1b + b1b ; atomically scatter h into sum[col].
// Per-wave tiling: 2 M-tiles x 2 N-tiles (2 A-frag + 2 B-frag loads per
// 4 WMMAs -> minimal LDS traffic per matrix op).
__global__ __launch_bounds__(256, 1)
void edge_mlp_kernel(const float* __restrict__ x,
                     const int* __restrict__ erow,
                     const int* __restrict__ ecol,
                     const float* __restrict__ eattr,
                     const _Float16* __restrict__ w1aT,
                     const _Float16* __restrict__ w1bT,
                     const float* __restrict__ b1a,
                     const float* __restrict__ b1b,
                     float* __restrict__ dsum,
                     float* __restrict__ dcnt,
                     int nE, int ntiles) {
    __shared__ __align__(16) _Float16 wA[256 * W1_STRIDE];   // 135168 B
    __shared__ __align__(16) _Float16 wB[256 * W1_STRIDE];   // 135168 B
    __shared__ __align__(16) _Float16 msg[32 * MSG_STRIDE];  // 16896 B
    __shared__ __align__(16) _Float16 h1[32 * MSG_STRIDE];   // 16896 B

    const int tid = threadIdx.x;

    // cooperative weight load into padded LDS (once per block)
    {
        const v8h* srcA = (const v8h*)w1aT;
        const v8h* srcB = (const v8h*)w1bT;
        for (int i = tid; i < 8192; i += 256) {   // 8192 v8h = 256x256 f16
            int n = i >> 5, kk = (i & 31) << 3;
            *(v8h*)&wA[n * W1_STRIDE + kk] = srcA[i];
            *(v8h*)&wB[n * W1_STRIDE + kk] = srcB[i];
        }
    }

    const int lane  = tid & 31;
    const int wave  = tid >> 5;
    const int lrow  = lane & 15;
    const int khalf = (lane >> 4) << 3;   // 0 or 8
    const int ntB   = wave << 1;          // 2 N tiles per wave (both M tiles)

    float bias1[2], bias2[2];
#pragma unroll
    for (int t = 0; t < 2; ++t) {
        int c = (ntB + t) * 16 + lrow;
        bias1[t] = b1a[c];
        bias2[t] = b1b[c];
    }

    const int et   = tid >> 3;   // edge-in-tile for gather (0..31)
    const int part = tid & 7;    // 32-float chunk (0..7)

    const v8f vzero = {0.f,0.f,0.f,0.f,0.f,0.f,0.f,0.f};

    for (int tile = blockIdx.x; tile < ntiles; tile += gridDim.x) {
        const int ebase = tile * 32;
        __syncthreads();   // previous tile fully consumed

        // per-edge in-degree count
        if (tid < 32 && (ebase + tid) < nE)
            unsafeAtomicAdd(&dcnt[ecol[ebase + tid]], 1.0f);

        // gather msg rows (f32 -> f16 LDS)
        {
            int e = ebase + et;
            if (e >= nE) e = nE - 1;
            const float* src;
            if (part < 4)
                src = x + (long)erow[e] * 128 + part * 32;
            else
                src = eattr + (long)e * 128 + (part - 4) * 32;
            _Float16* dst = &msg[et * MSG_STRIDE + part * 32];
#pragma unroll
            for (int j = 0; j < 4; ++j) {
                v4f a = ((const v4f*)src)[2 * j];
                v4f b = ((const v4f*)src)[2 * j + 1];
                v8h hv;
                hv[0]=(_Float16)a[0]; hv[1]=(_Float16)a[1]; hv[2]=(_Float16)a[2]; hv[3]=(_Float16)a[3];
                hv[4]=(_Float16)b[0]; hv[5]=(_Float16)b[1]; hv[6]=(_Float16)b[2]; hv[7]=(_Float16)b[3];
                *(v8h*)(dst + 8 * j) = hv;
            }
        }
        __syncthreads();

        // ---- GEMM1: [32,256] x [256,256]
        v8f acc00 = vzero, acc01 = vzero, acc10 = vzero, acc11 = vzero;
        {
            const _Float16* arow0 = &msg[lrow * MSG_STRIDE];
            const _Float16* arow1 = &msg[(16 + lrow) * MSG_STRIDE];
            const _Float16* brow0 = &wA[(ntB * 16 + lrow) * W1_STRIDE];
            const _Float16* brow1 = &wA[((ntB + 1) * 16 + lrow) * W1_STRIDE];
#pragma unroll
            for (int kc = 0; kc < 8; ++kc) {
                int k0 = kc * 32 + khalf;
                v16h a0 = load_frag(arow0, k0);
                v16h a1 = load_frag(arow1, k0);
                v16h b0 = load_frag(brow0, k0);
                v16h b1 = load_frag(brow1, k0);
                acc00 = WMMA_F16(a0, b0, acc00);
                acc10 = WMMA_F16(a1, b0, acc10);
                acc01 = WMMA_F16(a0, b1, acc01);
                acc11 = WMMA_F16(a1, b1, acc11);
            }
        }
        // bias + ELU -> h1 (f16 LDS)
        {
            int c0 = ntB * 16 + lrow;
            int c1 = c0 + 16;
#pragma unroll
            for (int r = 0; r < 8; ++r) {
                int r0 = khalf + r;          // M tile 0 rows
                int r1 = 16 + khalf + r;     // M tile 1 rows
                h1[r0 * MSG_STRIDE + c0] = (_Float16)elu_f(acc00[r] + bias1[0]);
                h1[r0 * MSG_STRIDE + c1] = (_Float16)elu_f(acc01[r] + bias1[1]);
                h1[r1 * MSG_STRIDE + c0] = (_Float16)elu_f(acc10[r] + bias1[0]);
                h1[r1 * MSG_STRIDE + c1] = (_Float16)elu_f(acc11[r] + bias1[1]);
            }
        }
        __syncthreads();

        // ---- GEMM2: [32,256] x [256,256]
        acc00 = vzero; acc01 = vzero; acc10 = vzero; acc11 = vzero;
        {
            const _Float16* arow0 = &h1[lrow * MSG_STRIDE];
            const _Float16* arow1 = &h1[(16 + lrow) * MSG_STRIDE];
            const _Float16* brow0 = &wB[(ntB * 16 + lrow) * W1_STRIDE];
            const _Float16* brow1 = &wB[((ntB + 1) * 16 + lrow) * W1_STRIDE];
#pragma unroll
            for (int kc = 0; kc < 8; ++kc) {
                int k0 = kc * 32 + khalf;
                v16h a0 = load_frag(arow0, k0);
                v16h a1 = load_frag(arow1, k0);
                v16h b0 = load_frag(brow0, k0);
                v16h b1 = load_frag(brow1, k0);
                acc00 = WMMA_F16(a0, b0, acc00);
                acc10 = WMMA_F16(a1, b0, acc10);
                acc01 = WMMA_F16(a0, b1, acc01);
                acc11 = WMMA_F16(a1, b1, acc11);
            }
        }
        // scatter h + b1b into dsum[col]
        {
            int nodes0[8], nodes1[8];
            bool val0[8], val1[8];
#pragma unroll
            for (int r = 0; r < 8; ++r) {
                int e0 = ebase + khalf + r;
                int e1 = ebase + 16 + khalf + r;
                val0[r] = e0 < nE;  nodes0[r] = ecol[val0[r] ? e0 : 0];
                val1[r] = e1 < nE;  nodes1[r] = ecol[val1[r] ? e1 : 0];
            }
            int c0 = ntB * 16 + lrow;
            int c1 = c0 + 16;
#pragma unroll
            for (int r = 0; r < 8; ++r) {
                if (val0[r]) {
                    unsafeAtomicAdd(&dsum[(long)nodes0[r] * 256 + c0], acc00[r] + bias2[0]);
                    unsafeAtomicAdd(&dsum[(long)nodes0[r] * 256 + c1], acc01[r] + bias2[1]);
                }
                if (val1[r]) {
                    unsafeAtomicAdd(&dsum[(long)nodes1[r] * 256 + c0], acc10[r] + bias2[0]);
                    unsafeAtomicAdd(&dsum[(long)nodes1[r] * 256 + c1], acc11[r] + bias2[1]);
                }
            }
        }
    }
}

// ---------------------------------------------------------------- node MLP
// z = [x || sum/max(cnt,1)]  (K=384); out = ELU(z@W2a+b2a) @ W2b + b2b
__global__ __launch_bounds__(256, 1)
void node_mlp_kernel(const float* __restrict__ x,
                     const float* __restrict__ dsum,
                     const float* __restrict__ dcnt,
                     const _Float16* __restrict__ w2aT,
                     const _Float16* __restrict__ w2bT,
                     const float* __restrict__ b2a,
                     const float* __restrict__ b2b,
                     float* __restrict__ out,
                     int nN, int ntiles) {
    __shared__ __align__(16) _Float16 wA[256 * W2A_STRIDE];  // 200704 B
    __shared__ __align__(16) _Float16 wB[128 * W2B_STRIDE];  // 67584 B
    __shared__ __align__(16) _Float16 z[32 * Z_STRIDE];      // 25088 B
    __shared__ __align__(16) _Float16 h[32 * MSG_STRIDE];    // 16896 B

    const int tid = threadIdx.x;
    {
        const v8h* srcA = (const v8h*)w2aT;   // 12288 v8h (256 x 384)
        for (int i = tid; i < 12288; i += 256) {
            int n = i / 48, kk = (i - n * 48) << 3;
            *(v8h*)&wA[n * W2A_STRIDE + kk] = srcA[i];
        }
        const v8h* srcB = (const v8h*)w2bT;   // 4096 v8h (128 x 256)
        for (int i = tid; i < 4096; i += 256) {
            int n = i >> 5, kk = (i & 31) << 3;
            *(v8h*)&wB[n * W2B_STRIDE + kk] = srcB[i];
        }
    }

    const int lane  = tid & 31;
    const int wave  = tid >> 5;
    const int lrow  = lane & 15;
    const int khalf = (lane >> 4) << 3;
    const int ntB   = wave << 1;        // GEMM-a: 2 N tiles/wave, both M tiles
    const int mt2   = wave >> 2;        // GEMM-b: 1 M tile
    const int nt2B  = (wave & 3) << 1;  // GEMM-b: 2 N tiles/wave

    float biasa[2], biasb[2];
#pragma unroll
    for (int t = 0; t < 2; ++t) biasa[t] = b2a[(ntB + t) * 16 + lrow];
#pragma unroll
    for (int t = 0; t < 2; ++t) biasb[t] = b2b[(nt2B + t) * 16 + lrow];

    const int et   = tid >> 3;  // node-in-tile (0..31)
    const int part = tid & 7;

    const v8f vzero = {0.f,0.f,0.f,0.f,0.f,0.f,0.f,0.f};

    for (int tile = blockIdx.x; tile < ntiles; tile += gridDim.x) {
        const int nbase = tile * 32;
        __syncthreads();

        // build z rows
        {
            int node = nbase + et;
            bool ok = node < nN;
            int nsafe = ok ? node : 0;
            // x part: 16 cols each
            {
                const float* src = x + (long)nsafe * 128 + part * 16;
                _Float16* dst = &z[et * Z_STRIDE + part * 16];
#pragma unroll
                for (int j = 0; j < 2; ++j) {
                    v4f a = ((const v4f*)src)[2 * j];
                    v4f b = ((const v4f*)src)[2 * j + 1];
                    v8h hv;
                    if (ok) {
                        hv[0]=(_Float16)a[0]; hv[1]=(_Float16)a[1]; hv[2]=(_Float16)a[2]; hv[3]=(_Float16)a[3];
                        hv[4]=(_Float16)b[0]; hv[5]=(_Float16)b[1]; hv[6]=(_Float16)b[2]; hv[7]=(_Float16)b[3];
                    } else {
                        hv = (v8h){0,0,0,0,0,0,0,0};
                    }
                    *(v8h*)(dst + 8 * j) = hv;
                }
            }
            // agg part: 32 cols each, scaled by 1/max(cnt,1)
            {
                float c = dcnt[nsafe];
                float inv = ok ? (1.0f / fmaxf(c, 1.0f)) : 0.0f;
                const float* src = dsum + (long)nsafe * 256 + part * 32;
                _Float16* dst = &z[et * Z_STRIDE + 128 + part * 32];
#pragma unroll
                for (int j = 0; j < 4; ++j) {
                    v4f a = ((const v4f*)src)[2 * j];
                    v4f b = ((const v4f*)src)[2 * j + 1];
                    v8h hv;
                    hv[0]=(_Float16)(a[0]*inv); hv[1]=(_Float16)(a[1]*inv);
                    hv[2]=(_Float16)(a[2]*inv); hv[3]=(_Float16)(a[3]*inv);
                    hv[4]=(_Float16)(b[0]*inv); hv[5]=(_Float16)(b[1]*inv);
                    hv[6]=(_Float16)(b[2]*inv); hv[7]=(_Float16)(b[3]*inv);
                    *(v8h*)(dst + 8 * j) = hv;
                }
            }
        }
        __syncthreads();

        // ---- GEMM-a: [32,384] x [384,256]
        v8f acc00 = vzero, acc01 = vzero, acc10 = vzero, acc11 = vzero;
        {
            const _Float16* arow0 = &z[lrow * Z_STRIDE];
            const _Float16* arow1 = &z[(16 + lrow) * Z_STRIDE];
            const _Float16* brow0 = &wA[(ntB * 16 + lrow) * W2A_STRIDE];
            const _Float16* brow1 = &wA[((ntB + 1) * 16 + lrow) * W2A_STRIDE];
#pragma unroll
            for (int kc = 0; kc < 12; ++kc) {
                int k0 = kc * 32 + khalf;
                v16h a0 = load_frag(arow0, k0);
                v16h a1 = load_frag(arow1, k0);
                v16h b0 = load_frag(brow0, k0);
                v16h b1 = load_frag(brow1, k0);
                acc00 = WMMA_F16(a0, b0, acc00);
                acc10 = WMMA_F16(a1, b0, acc10);
                acc01 = WMMA_F16(a0, b1, acc01);
                acc11 = WMMA_F16(a1, b1, acc11);
            }
        }
        {
            int c0 = ntB * 16 + lrow;
            int c1 = c0 + 16;
#pragma unroll
            for (int r = 0; r < 8; ++r) {
                int r0 = khalf + r;
                int r1 = 16 + khalf + r;
                h[r0 * MSG_STRIDE + c0] = (_Float16)elu_f(acc00[r] + biasa[0]);
                h[r0 * MSG_STRIDE + c1] = (_Float16)elu_f(acc01[r] + biasa[1]);
                h[r1 * MSG_STRIDE + c0] = (_Float16)elu_f(acc10[r] + biasa[0]);
                h[r1 * MSG_STRIDE + c1] = (_Float16)elu_f(acc11[r] + biasa[1]);
            }
        }
        __syncthreads();

        // ---- GEMM-b: [32,256] x [256,128]
        v8f accb0 = vzero, accb1 = vzero;
        {
            const _Float16* arow = &h[(mt2 * 16 + lrow) * MSG_STRIDE];
            const _Float16* brow0 = &wB[(nt2B * 16 + lrow) * W2B_STRIDE];
            const _Float16* brow1 = &wB[((nt2B + 1) * 16 + lrow) * W2B_STRIDE];
#pragma unroll
            for (int kc = 0; kc < 8; ++kc) {
                int k0 = kc * 32 + khalf;
                v16h a = load_frag(arow, k0);
                v16h b0 = load_frag(brow0, k0);
                v16h b1 = load_frag(brow1, k0);
                accb0 = WMMA_F16(a, b0, accb0);
                accb1 = WMMA_F16(a, b1, accb1);
            }
        }
        // store output f32
        {
            int c0 = nt2B * 16 + lrow;
            int c1 = c0 + 16;
#pragma unroll
            for (int r = 0; r < 8; ++r) {
                int row = nbase + mt2 * 16 + khalf + r;
                if (row < nN) {
                    out[(long)row * 128 + c0] = accb0[r] + biasb[0];
                    out[(long)row * 128 + c1] = accb1[r] + biasb[1];
                }
            }
        }
    }
}

// ---------------------------------------------------------------- launch
extern "C" void kernel_launch(void* const* d_in, const int* in_sizes, int n_in,
                              void* d_out, int out_size, void* d_ws, size_t ws_size,
                              hipStream_t stream) {
    const float* x     = (const float*)d_in[0];
    const int*   eidx  = (const int*)d_in[1];   // [2, E] flat
    const float* eattr = (const float*)d_in[2];
    // d_in[3] = u, d_in[4] = batch : unused by reference node model
    const float* W1a = (const float*)d_in[5];
    const float* b1a = (const float*)d_in[6];
    const float* W1b = (const float*)d_in[7];
    const float* b1b = (const float*)d_in[8];
    const float* W2a = (const float*)d_in[9];
    const float* b2a = (const float*)d_in[10];
    const float* W2b = (const float*)d_in[11];
    const float* b2b = (const float*)d_in[12];

    const int N = in_sizes[0] / 128;
    const int E = in_sizes[2] / 128;

    char* ws = (char*)d_ws;
    _Float16* w1aT = (_Float16*)(ws);                 // 256*256*2 = 131072
    _Float16* w1bT = (_Float16*)(ws + 131072);        // 131072
    _Float16* w2aT = (_Float16*)(ws + 262144);        // 256*384*2 = 196608
    _Float16* w2bT = (_Float16*)(ws + 458752);        // 128*256*2 = 65536
    float*    dsum = (float*)(ws + 524288);           // N*256*4
    float*    dcnt = dsum + (long)N * 256;            // N*4

    convert_weights_kernel<<<256, 256, 0, stream>>>(W1a, W1b, W2a, W2b,
                                                    w1aT, w1bT, w2aT, w2bT);
    zero_kernel<<<2048, 256, 0, stream>>>(dsum, (long)N * 256 + N);

    const int etiles = (E + 31) / 32;
    int eblocks = etiles < 1024 ? etiles : 1024;
    edge_mlp_kernel<<<eblocks, 256, 0, stream>>>(x, eidx, eidx + E, eattr,
                                                 w1aT, w1bT, b1a, b1b,
                                                 dsum, dcnt, E, etiles);

    const int ntilesN = (N + 31) / 32;
    int nblocks = ntilesN < 512 ? ntilesN : 512;
    node_mlp_kernel<<<nblocks, 256, 0, stream>>>(x, dsum, dcnt,
                                                 w2aT, w2bT, b2a, b2b,
                                                 (float*)d_out, N, ntilesN);
}
